// FrequencyLayer_85847806313214
// MI455X (gfx1250) — matrix operand order repaired and checked by reading, not compiled, not run on hardware
//
#include <hip/hip_runtime.h>

typedef __attribute__((ext_vector_type(16))) _Float16 v16h;
typedef __attribute__((ext_vector_type(8)))  _Float16 v8h;
typedef __attribute__((ext_vector_type(8)))  float    v8f;

#define SDIM 256
#define IMG  (SDIM * SDIM)
#define NIMG 512   // B=16 * C=32

#define WMMA_F16(A, B, C) \
  __builtin_amdgcn_wmma_f32_16x16x32_f16(false, (A), false, (B), (short)0, (C), false, false)

// ---------------------------------------------------------------------------
// Generate f16 DFT bases in workspace: Cm[q][r]=cos(2*pi*q*r/S), Sm=sin(...).
// Angle is periodic in (q*r mod S): reduce exactly in integers first.
// ---------------------------------------------------------------------------
__global__ void gen_bases_kernel(_Float16* __restrict__ cm,
                                 _Float16* __restrict__ sm) {
  int idx = blockIdx.x * blockDim.x + threadIdx.x;   // 65536 threads
  int q = idx >> 8;
  int r = idx & 255;
  int m = (q * r) & 255;                              // exact reduction mod S
  float ang = (float)m * 0.0245436926f;               // 2*pi/256
  cm[idx] = (_Float16)__cosf(ang);
  sm[idx] = (_Float16)__sinf(ang);
}

// ---------------------------------------------------------------------------
// Fragment loaders (wave32 WMMA operand layouts, 16x16x32 f16)
// ---------------------------------------------------------------------------

// A fragment from a global fp32 row: elems 0..7 = row[kb..kb+7],
// elems 8..15 = row[kb+16..kb+23], converted to f16. (kb = k0 + (lane>=16)*8)
__device__ __forceinline__ v16h load_a_f32(const float* __restrict__ row, int kb) {
  float4 x0 = *(const float4*)(row + kb);
  float4 x1 = *(const float4*)(row + kb + 4);
  float4 x2 = *(const float4*)(row + kb + 16);
  float4 x3 = *(const float4*)(row + kb + 20);
  v16h a;
  a[0]  = (_Float16)x0.x; a[1]  = (_Float16)x0.y;
  a[2]  = (_Float16)x0.z; a[3]  = (_Float16)x0.w;
  a[4]  = (_Float16)x1.x; a[5]  = (_Float16)x1.y;
  a[6]  = (_Float16)x1.z; a[7]  = (_Float16)x1.w;
  a[8]  = (_Float16)x2.x; a[9]  = (_Float16)x2.y;
  a[10] = (_Float16)x2.z; a[11] = (_Float16)x2.w;
  a[12] = (_Float16)x3.x; a[13] = (_Float16)x3.y;
  a[14] = (_Float16)x3.z; a[15] = (_Float16)x3.w;
  return a;
}

// B fragment from a symmetric f16 matrix (row n == column n): elems 0..7 =
// row[kb..kb+7], 8..15 = row[kb+16..kb+23], kb = k0 + (lane>=16)*8 -> two b128 loads.
__device__ __forceinline__ v16h load_b_f16(const _Float16* __restrict__ mat,
                                           int n, int kb) {
  union { v16h v; uint4 u[2]; } t;
  const uint4* p = (const uint4*)(mat + n * SDIM + kb);
  t.u[0] = p[0];
  t.u[1] = p[2];
  return t.v;
}

// A fragment from an LDS f16 row (elems 0..7 = row[kb..], 8..15 = row[kb+16..])
__device__ __forceinline__ v16h load_a_lds(const _Float16* row, int kb) {
  union { v16h v; uint4 u[2]; } t;
  t.u[0] = *(const uint4*)(row + kb);
  t.u[1] = *(const uint4*)(row + kb + 16);
  return t.v;
}

// ---------------------------------------------------------------------------
// Fused per-image 2-D DFT:
//   stage 1: T1 = X*Cm, T2 = X*Sm  -> stored TRANSPOSED as f16 in LDS
//   stage 2: re = T1^T*Cm - T2^T*Sm ; im = -(T2^T*Cm + T1^T*Sm)
// One workgroup per image; 8 wave32s; 256 KB dynamic LDS (fits 320 KB WGP LDS).
// Stage 1 amortizes one fp32->f16 A-fragment over 4 tn tiles (8 WMMAs);
// stage 2 amortizes one LDS A-fragment pair over 2 ts tiles (8 WMMAs).
// ---------------------------------------------------------------------------
__global__ __launch_bounds__(256, 1)
void dft2d_wmma_kernel(const float* __restrict__ x,
                       const _Float16* __restrict__ cm,
                       const _Float16* __restrict__ sm,
                       float* __restrict__ out_re,
                       float* __restrict__ out_im) {
  extern __shared__ _Float16 lds[];
  _Float16* U1 = lds;        // T1^T : [r][p], 256x256 f16 (128 KB)
  _Float16* U2 = lds + IMG;  // T2^T : [r][p]               (128 KB)

  const int tid  = threadIdx.x;
  const int wave = tid >> 5;
  const int lane = tid & 31;
  const int ln16 = lane & 15;
  const int hi   = lane >> 4;

  const float* X = x + (size_t)blockIdx.x * IMG;

  // ---- Stage 1: 64 groups of 4 tn-tiles; 8 waves x 8 groups ----
  for (int g = wave; g < 64; g += 8) {
    const int tm  = g >> 2;         // p block (rows of X)
    const int tnb = (g & 3) << 2;   // first of 4 consecutive r blocks
    v8f ac[4] = {v8f{}, v8f{}, v8f{}, v8f{}};   // X*Cm tiles
    v8f as[4] = {v8f{}, v8f{}, v8f{}, v8f{}};   // X*Sm tiles
    const float* arow = X + (tm * 16 + ln16) * SDIM;
    const int bn0 = tnb * 16 + ln16;
    for (int k0 = 0; k0 < SDIM; k0 += 32) {
      const v16h a   = load_a_f32(arow, k0 + hi * 8);  // converted once,
      const int  kbB = k0 + hi * 8;                    // feeds 8 WMMAs
      #pragma unroll
      for (int j = 0; j < 4; ++j) {
        v16h bc = load_b_f16(cm, bn0 + j * 16, kbB);
        v16h bs = load_b_f16(sm, bn0 + j * 16, kbB);
        ac[j] = WMMA_F16(a, bc, ac[j]);
        as[j] = WMMA_F16(a, bs, as[j]);
      }
    }
    // D layout: lane holds N=ln16, M = hi*8 + v (consecutive) -> transposed
    // store into U[r][p] is one contiguous b128 per lane after f16 packing.
    const int col = tm * 16 + hi * 8;   // p
    #pragma unroll
    for (int j = 0; j < 4; ++j) {
      const int row = (tnb + j) * 16 + ln16;   // r
      union { v8h h; uint4 u; } p1, p2;
      #pragma unroll
      for (int i = 0; i < 8; ++i) {
        p1.h[i] = (_Float16)ac[j][i];
        p2.h[i] = (_Float16)as[j][i];
      }
      *(uint4*)(U1 + row * SDIM + col) = p1.u;
      *(uint4*)(U2 + row * SDIM + col) = p2.u;
    }
  }

  __syncthreads();

  float* ore = out_re + (size_t)blockIdx.x * IMG;
  float* oim = out_im + (size_t)blockIdx.x * IMG;

  // ---- Stage 2: 128 groups of 2 ts-tiles; 8 waves x 16 groups ----
  for (int g = wave; g < 128; g += 8) {
    const int tr  = g >> 3;         // r block (rows of U)
    const int tsb = (g & 7) << 1;   // first of 2 consecutive s blocks
    v8f cc[2] = {v8f{}, v8f{}};
    v8f cs[2] = {v8f{}, v8f{}};
    v8f sc[2] = {v8f{}, v8f{}};
    v8f ss[2] = {v8f{}, v8f{}};
    const _Float16* a1row = U1 + (tr * 16 + ln16) * SDIM;
    const _Float16* a2row = U2 + (tr * 16 + ln16) * SDIM;
    const int bn0 = tsb * 16 + ln16;
    for (int k0 = 0; k0 < SDIM; k0 += 32) {
      const v16h a1  = load_a_lds(a1row, k0 + hi * 8);  // shared by 2 tiles
      const v16h a2  = load_a_lds(a2row, k0 + hi * 8);
      const int  kbB = k0 + hi * 8;
      #pragma unroll
      for (int j = 0; j < 2; ++j) {
        v16h bc = load_b_f16(cm, bn0 + j * 16, kbB);
        v16h bs = load_b_f16(sm, bn0 + j * 16, kbB);
        cc[j] = WMMA_F16(a1, bc, cc[j]);
        ss[j] = WMMA_F16(a2, bs, ss[j]);
        sc[j] = WMMA_F16(a2, bc, sc[j]);
        cs[j] = WMMA_F16(a1, bs, cs[j]);
      }
    }
    // re = cc - ss ; im = -(sc + cs). Output is write-once streaming data
    // (256 MB > 192 MB L2): store non-temporal to keep bases/X resident.
    const int rbase = tr * 16 + hi * 8;
    #pragma unroll
    for (int j = 0; j < 2; ++j) {
      const int scol = (tsb + j) * 16 + ln16;
      #pragma unroll
      for (int v = 0; v < 8; ++v) {
        __builtin_nontemporal_store(cc[j][v] - ss[j][v],
                                    ore + (rbase + v) * SDIM + scol);
        __builtin_nontemporal_store(-(sc[j][v] + cs[j][v]),
                                    oim + (rbase + v) * SDIM + scol);
      }
    }
  }
}

// ---------------------------------------------------------------------------
extern "C" void kernel_launch(void* const* d_in, const int* in_sizes, int n_in,
                              void* d_out, int out_size, void* d_ws, size_t ws_size,
                              hipStream_t stream) {
  const float* x = (const float*)d_in[0];          // [16,32,256,256] fp32

  _Float16* cmv = (_Float16*)d_ws;                 // 128 KB
  _Float16* smv = cmv + IMG;                       // 128 KB
  float* out_re = (float*)d_out;                   // [16,32,256,256]
  float* out_im = out_re + (size_t)NIMG * IMG;     // second tuple element

  const int lds_bytes = 2 * IMG * (int)sizeof(_Float16);  // 256 KB
  hipFuncSetAttribute(reinterpret_cast<const void*>(dft2d_wmma_kernel),
                      hipFuncAttributeMaxDynamicSharedMemorySize, lds_bytes);

  gen_bases_kernel<<<SDIM, SDIM, 0, stream>>>(cmv, smv);
  dft2d_wmma_kernel<<<NIMG, 256, lds_bytes, stream>>>(x, cmv, smv, out_re, out_im);
}
